// GGNN_16080357556806
// MI455X (gfx1250) — matrix operand order repaired and checked by reading, not compiled, not run on hardware
//
#include <hip/hip_runtime.h>
#include <hip/hip_bf16.h>

// ---------------------------------------------------------------------------
// Types for CDNA5 WMMA (wave32): A/B are 16 bf16 per lane (8 VGPRs),
// C/D are 8 f32 per lane (8 VGPRs).
// ---------------------------------------------------------------------------
typedef __attribute__((ext_vector_type(16))) __bf16 v16bf;
typedef __attribute__((ext_vector_type(8)))  float  v8f;

union Frag { v16bf v; unsigned int u[8]; };
union Acc  { v8f   v; float f[8]; };

#define N_NODE 200000
#define DIM    64
#define F_DIM  32
#define F_OUT  64
#define HIDE   128
#define BATCH  256
#define SEQ    50

__device__ __forceinline__ unsigned short f2bf(float f) {
    unsigned u = __float_as_uint(f);
    unsigned r = u + 0x7fffu + ((u >> 16) & 1u);   // round-to-nearest-even
    return (unsigned short)(r >> 16);
}
__device__ __forceinline__ unsigned pack2bf(float lo, float hi) {
    return (unsigned)f2bf(lo) | ((unsigned)f2bf(hi) << 16);
}
__device__ __forceinline__ float bf2f(unsigned short h) {
    return __uint_as_float(((unsigned)h) << 16);
}

// Cheap transcendentals: v_exp_f32 + v_rcp_f32, no IEEE-div chain.
__device__ __forceinline__ float fast_sigmoid(float x) {
    return __builtin_amdgcn_rcpf(1.0f + __expf(-x));
}
__device__ __forceinline__ float fast_tanh(float x) {
    float e = __expf(2.0f * x);
    return (e - 1.0f) * __builtin_amdgcn_rcpf(e + 1.0f);
}

__device__ __forceinline__ v8f wmma_bf16(const Frag& a, const Frag& b, v8f c) {
    return __builtin_amdgcn_wmma_f32_16x16x32_bf16(
        /*neg_a=*/false, a.v, /*neg_b=*/false, b.v,
        /*c_mod=*/(short)0, c, /*reuse_a=*/false, /*reuse_b=*/false);
}

// A-matrix 16x32 bf16 layout (ISA 7.12.2): lane holds row M = lane%16;
// VGPR v holds K-pair K0 = ((v<4)?0:16) + 8*half + 2*(v&3).
__device__ __forceinline__ Frag load_a_bf16(const unsigned short* rowptr,
                                            int kbase, int g /*lane>>4*/) {
    Frag f;
#pragma unroll
    for (int v = 0; v < 8; ++v) {
        int k0 = ((v < 4) ? 0 : 16) + g * 8 + 2 * (v & 3) + kbase;
        f.u[v] = *(const unsigned int*)(rowptr + k0);
    }
    return f;
}

__device__ __forceinline__ Frag load_a_f32(const float* rowptr,
                                           int kbase, int g) {
    Frag f;
#pragma unroll
    for (int v = 0; v < 8; ++v) {
        int k0 = ((v < 4) ? 0 : 16) + g * 8 + 2 * (v & 3) + kbase;
        f.u[v] = pack2bf(rowptr[k0], rowptr[k0 + 1]);
    }
    return f;
}

// B-matrix 32x16 bf16 (ISA 7.12.2): lane holds column N = lane%16; VGPR v
// holds K-pair K0 = ((v<4)?0:16) + 8*half + 2*(v&3). One B column == one contiguous bf16 row.
__device__ __forceinline__ Frag load_b_colrow_bf16(const unsigned short* colptr,
                                                   int kbase, int g) {
    Frag f;
#pragma unroll
    for (int v = 0; v < 8; ++v) {
        int k0 = ((v < 4) ? 0 : 16) + g * 8 + 2 * (v & 3) + kbase;
        f.u[v] = *(const unsigned int*)(colptr + k0);
    }
    return f;
}

__device__ __forceinline__ float wave_reduce_sum(float x) {
    x += __shfl_xor(x, 16, 32);
    x += __shfl_xor(x, 8, 32);
    x += __shfl_xor(x, 4, 32);
    x += __shfl_xor(x, 2, 32);
    x += __shfl_xor(x, 1, 32);
    return x;
}

// ---------------------------------------------------------------------------
// Kernel 0: zero per-launch accumulators (graph-replay safe).
// ---------------------------------------------------------------------------
__global__ void k_init(unsigned* rowmax, float* rowsum, float* scal) {
    int i = threadIdx.x;
    if (i < BATCH) { rowmax[i] = 0u; rowsum[i] = 0.0f; }
    if (i < 8) scal[i] = 0.0f;
}

// ---------------------------------------------------------------------------
// Kernel 0b: transpose weights to bf16 column-major (Wt[n][k]) so every B
// fragment in the GEMM kernels becomes 8 aligned dword loads (no f32->bf16
// conversion VALU in the hot loops). Total ~132 KB, L2 resident.
// ---------------------------------------------------------------------------
__global__ __launch_bounds__(256) void k_prep(
    const float* __restrict__ f_w, const float* __restrict__ Wsm,
    const float* __restrict__ Wg, const float* __restrict__ Wc,
    unsigned short* __restrict__ f_wT, unsigned short* __restrict__ WsmT,
    unsigned short* __restrict__ WgHT, unsigned short* __restrict__ WcHT) {
    int i = blockIdx.x * 256 + threadIdx.x;        // grid covers 32768
    if (i < F_OUT * F_DIM) {                       // f_wT: [64][32]
        int n = i >> 5, k = i & 31;
        f_wT[i] = f2bf(f_w[k * F_OUT + n]);
    }
    if (i < HIDE * HIDE) {                         // WsmT: [128][128]
        int n = i >> 7, k = i & 127;
        WsmT[i] = f2bf(Wsm[k * HIDE + n]);
    }
    if (i < 2 * HIDE * HIDE) {                     // WgHT: [256][128]
        int n = i >> 7, k = i & 127;
        WgHT[i] = f2bf(Wg[(size_t)(HIDE + k) * (2 * HIDE) + n]);
    }
    if (i < HIDE * HIDE) {                         // WcHT: [128][128]
        int n = i >> 7, k = i & 127;
        WcHT[i] = f2bf(Wc[(size_t)(HIDE + k) * HIDE + n]);
    }
}

// ---------------------------------------------------------------------------
// Kernel 1: fused  f_emb = sigmoid(other_feat@f_w),
//                  in_emb = sigmoid([emb|f_emb]@emb_smaller + b)  -> bf16
// Accumulates sum(f_embedding^2). One 16-row node tile per wave.
// ---------------------------------------------------------------------------
__global__ __launch_bounds__(128) void k_inemb(
    const float* __restrict__ emb, const float* __restrict__ ofeat,
    const unsigned short* __restrict__ f_wT,
    const unsigned short* __restrict__ WsmT, const float* __restrict__ bsm,
    unsigned short* __restrict__ inemb_bf, float* __restrict__ l2_acc) {
    __shared__ unsigned short smem[4][16][132];   // bf16 concat tile, padded
    int wave = threadIdx.x >> 5, lane = threadIdx.x & 31;
    int lane16 = lane & 15, g = lane >> 4;
    int node0 = (blockIdx.x * 4 + wave) * 16;
    float l2 = 0.0f;

    // stage raw embeddings (cols 0..63) as bf16 + L2 partial (float2 loads)
    for (int i = lane; i < 16 * (DIM / 2); i += 32) {
        int r = i >> 5, c = (i & 31) * 2;
        float2 x = *(const float2*)(emb + (size_t)(node0 + r) * DIM + c);
        l2 += x.x * x.x + x.y * x.y;
        *(unsigned*)&smem[wave][r][c] = pack2bf(x.x, x.y);
    }

    // f_emb GEMM: 16x32 @ 32x64, 4 N-tiles, K=32 (1 WMMA each)
    Frag a0 = load_a_f32(ofeat + (size_t)(node0 + lane16) * F_DIM, 0, g);
#pragma unroll
    for (int nt = 0; nt < 4; ++nt) {
        Acc acc;
#pragma unroll
        for (int i = 0; i < 8; ++i) acc.f[i] = 0.0f;
        Frag b = load_b_colrow_bf16(f_wT + (nt * 16 + lane16) * F_DIM, 0, g);
        acc.v = wmma_bf16(a0, b, acc.v);
#pragma unroll
        for (int v = 0; v < 8; ++v) {
            int M = v + 8 * g, N = nt * 16 + lane16;
            float s = fast_sigmoid(acc.f[v]);
            l2 += s * s;
            smem[wave][M][DIM + N] = f2bf(s);
        }
    }

    // main GEMM: 16x128 @ 128x128, 8 N-tiles x 4 K-chunks
#pragma unroll 1
    for (int nt = 0; nt < 8; ++nt) {
        Acc acc;
#pragma unroll
        for (int i = 0; i < 8; ++i) acc.f[i] = 0.0f;
        const unsigned short* bcol = WsmT + (nt * 16 + lane16) * HIDE;
#pragma unroll
        for (int kc = 0; kc < 4; ++kc) {
            Frag a = load_a_bf16(&smem[wave][lane16][0], kc * 32, g);
            Frag b = load_b_colrow_bf16(bcol, kc * 32, g);
            acc.v = wmma_bf16(a, b, acc.v);
        }
        float bias = bsm[nt * 16 + lane16];
#pragma unroll
        for (int v = 0; v < 8; ++v) {
            int M = v + 8 * g, N = nt * 16 + lane16;
            float s = fast_sigmoid(acc.f[v] + bias);
            inemb_bf[(size_t)(node0 + M) * HIDE + N] = f2bf(s);
        }
    }

    l2 = wave_reduce_sum(l2);
    if (lane == 0) atomicAdd(l2_acc, l2);
}

// ---------------------------------------------------------------------------
// Kernel 2: GRU step (x = 0):  gates = sigmoid(h@Wg_h + bg)
//   c = tanh((r*h)@Wc_h + bc);  h' = z*h + (1-z)*c
// Gathers h rows from bf16 in_emb, accumulates sum(h'^2), scatters u_emb rows.
// ---------------------------------------------------------------------------
__global__ __launch_bounds__(128) void k_gru(
    const unsigned short* __restrict__ inemb_bf, const int* __restrict__ item,
    const int* __restrict__ nsess, const unsigned short* __restrict__ WgHT,
    const float* __restrict__ bg, const unsigned short* __restrict__ WcHT,
    const float* __restrict__ bc, unsigned short* __restrict__ uemb_bf,
    float* __restrict__ l2_acc) {
    __shared__ unsigned short rh[4][16][132];
    __shared__ float zt[4][16][HIDE];
    int wave = threadIdx.x >> 5, lane = threadIdx.x & 31;
    int lane16 = lane & 15, g = lane >> 4;
    int row0 = (blockIdx.x * 4 + wave) * 16;
    int nid = item[row0 + lane16];                 // node id of this A-row
    const unsigned short* hrow = inemb_bf + (size_t)nid * HIDE;
    float l2 = 0.0f;

    // gates GEMM: 16x128 @ 128x256 (Wg_h = gru_gate_kernel rows 128..255)
#pragma unroll 1
    for (int nt = 0; nt < 16; ++nt) {
        Acc acc;
#pragma unroll
        for (int i = 0; i < 8; ++i) acc.f[i] = 0.0f;
        const unsigned short* bcol = WgHT + (nt * 16 + lane16) * HIDE;
#pragma unroll
        for (int kc = 0; kc < 4; ++kc) {
            Frag a = load_a_bf16(hrow, kc * 32, g);
            Frag b = load_b_colrow_bf16(bcol, kc * 32, g);
            acc.v = wmma_bf16(a, b, acc.v);
        }
        float bias = bg[nt * 16 + lane16];
#pragma unroll
        for (int v = 0; v < 8; ++v) {
            int M = v + 8 * g;
            float gate = fast_sigmoid(acc.f[v] + bias);
            int mnid = __shfl(nid, M, 32);
            if (nt < 8) {                          // r gate -> rh tile
                int N = nt * 16 + lane16;
                float h = bf2f(inemb_bf[(size_t)mnid * HIDE + N]);
                rh[wave][M][N] = f2bf(gate * h);
            } else {                               // z gate
                int N = (nt - 8) * 16 + lane16;
                zt[wave][M][N] = gate;
            }
        }
    }

    // candidate GEMM + combine (Wc_h = gru_cand_kernel rows 128..255)
#pragma unroll 1
    for (int nt = 0; nt < 8; ++nt) {
        Acc acc;
#pragma unroll
        for (int i = 0; i < 8; ++i) acc.f[i] = 0.0f;
        const unsigned short* bcol = WcHT + (nt * 16 + lane16) * HIDE;
#pragma unroll
        for (int kc = 0; kc < 4; ++kc) {
            Frag a = load_a_bf16(&rh[wave][lane16][0], kc * 32, g);
            Frag b = load_b_colrow_bf16(bcol, kc * 32, g);
            acc.v = wmma_bf16(a, b, acc.v);
        }
        float bias = bc[nt * 16 + lane16];
#pragma unroll
        for (int v = 0; v < 8; ++v) {
            int M = v + 8 * g, N = nt * 16 + lane16;
            float c = fast_tanh(acc.f[v] + bias);
            float z = zt[wave][M][N];
            int mnid = __shfl(nid, M, 32);
            float h = bf2f(inemb_bf[(size_t)mnid * HIDE + N]);
            float hn = z * h + (1.0f - z) * c;
            l2 += hn * hn;
            int grow = row0 + M;
            int b_ = grow / SEQ, s_ = grow % SEQ;
            if (s_ == nsess[b_] - 1)
                uemb_bf[(size_t)b_ * HIDE + N] = f2bf(hn);
        }
    }

    l2 = wave_reduce_sum(l2);
    if (lane == 0) atomicAdd(l2_acc, l2);
}

// ---------------------------------------------------------------------------
// Kernel 3: logits = u_emb @ in_emb^T  (256x128 @ 128x200000) + row maxima.
// Each wave: one 16-row strip, 25 sequential N-tiles, register A-frags.
// ---------------------------------------------------------------------------
__global__ __launch_bounds__(128) void k_logits(
    const unsigned short* __restrict__ uemb_bf,
    const unsigned short* __restrict__ inemb_bf,
    float* __restrict__ out_logits, unsigned* __restrict__ rowmax) {
    int wave = threadIdx.x >> 5, lane = threadIdx.x & 31;
    int lane16 = lane & 15, g = lane >> 4;
    int mt = blockIdx.y;                           // 0..15 (M-tile)
    int wc = blockIdx.x * 4 + wave;                // 0..499 (wave column)
    const unsigned short* arow = uemb_bf + (size_t)(mt * 16 + lane16) * HIDE;
    Frag a[4];
#pragma unroll
    for (int kc = 0; kc < 4; ++kc) a[kc] = load_a_bf16(arow, kc * 32, g);

    float rmax[8];
#pragma unroll
    for (int v = 0; v < 8; ++v) rmax[v] = -3.4e38f;

    for (int t = 0; t < 25; ++t) {
        int col0 = (wc * 25 + t) * 16;
        const unsigned short* brow = inemb_bf + (size_t)(col0 + lane16) * HIDE;
        if (t < 24)                                // global_prefetch_b8 next B row
            __builtin_prefetch(brow + 16 * HIDE, 0, 0);
        Acc acc;
#pragma unroll
        for (int i = 0; i < 8; ++i) acc.f[i] = 0.0f;
#pragma unroll
        for (int kc = 0; kc < 4; ++kc) {
            Frag b = load_b_colrow_bf16(brow, kc * 32, g);
            acc.v = wmma_bf16(a[kc], b, acc.v);
        }
#pragma unroll
        for (int v = 0; v < 8; ++v) {
            int M = mt * 16 + v + 8 * g;
            out_logits[(size_t)M * N_NODE + col0 + lane16] = acc.f[v];
            rmax[v] = fmaxf(rmax[v], acc.f[v]);
        }
    }

#pragma unroll
    for (int v = 0; v < 8; ++v) {
        float m = rmax[v];
        m = fmaxf(m, __shfl_xor(m, 1, 32));
        m = fmaxf(m, __shfl_xor(m, 2, 32));
        m = fmaxf(m, __shfl_xor(m, 4, 32));
        m = fmaxf(m, __shfl_xor(m, 8, 32));
        if (lane16 == 0) {
            unsigned u = __float_as_uint(m);
            unsigned key = u ^ (unsigned)(((int)u >> 31) | (int)0x80000000);
            atomicMax(&rowmax[mt * 16 + v + 8 * g], key);
        }
    }
}

// ---------------------------------------------------------------------------
// Kernel 4: exp(l - max) in place + per-row sums.
// ---------------------------------------------------------------------------
__global__ __launch_bounds__(256) void k_expsum(float* __restrict__ out_logits,
                                                const unsigned* __restrict__ rowmax,
                                                float* __restrict__ rowsum) {
    int row = blockIdx.y;
    unsigned key = rowmax[row];
    float m = __uint_as_float((key & 0x80000000u) ? (key ^ 0x80000000u) : ~key);
    size_t base = (size_t)row * N_NODE;
    int c0 = blockIdx.x * 2048 + threadIdx.x;
    float s = 0.0f;
#pragma unroll
    for (int i = 0; i < 8; ++i) {
        int c = c0 + i * 256;
        if (c < N_NODE) {
            float e = __expf(out_logits[base + c] - m);
            out_logits[base + c] = e;
            s += e;
        }
    }
    s = wave_reduce_sum(s);
    __shared__ float sh[8];
    if ((threadIdx.x & 31) == 0) sh[threadIdx.x >> 5] = s;
    __syncthreads();
    if (threadIdx.x == 0) {
        float t = 0.0f;
        for (int i = 0; i < 8; ++i) t += sh[i];
        atomicAdd(&rowsum[row], t);
    }
}

// ---------------------------------------------------------------------------
// Kernel 5: cross-entropy from exp-values (runs between expsum and norm).
// ---------------------------------------------------------------------------
__global__ void k_ce(const float* __restrict__ out_logits,
                     const float* __restrict__ rowsum,
                     const int* __restrict__ tar, float* __restrict__ ce) {
    int b = threadIdx.x;                           // 256 threads
    float e = out_logits[(size_t)b * N_NODE + tar[b]];
    float v = -(__logf(e) - __logf(rowsum[b]));
    v = wave_reduce_sum(v);
    __shared__ float sh[8];
    if ((threadIdx.x & 31) == 0) sh[threadIdx.x >> 5] = v;
    __syncthreads();
    if (threadIdx.x == 0) {
        float t = 0.0f;
        for (int i = 0; i < 8; ++i) t += sh[i];
        *ce = t;
    }
}

// ---------------------------------------------------------------------------
// Kernel 6: normalize -> softmax in place.
// ---------------------------------------------------------------------------
__global__ __launch_bounds__(256) void k_norm(float* __restrict__ out_logits,
                                              const float* __restrict__ rowsum) {
    int row = blockIdx.y;
    float inv = 1.0f / rowsum[row];
    size_t base = (size_t)row * N_NODE;
    int c0 = blockIdx.x * 2048 + threadIdx.x;
#pragma unroll
    for (int i = 0; i < 8; ++i) {
        int c = c0 + i * 256;
        if (c < N_NODE) out_logits[base + c] *= inv;
    }
}

// ---------------------------------------------------------------------------
// Kernel 7: sum(emb_smaller^2) + final loss.
// ---------------------------------------------------------------------------
__global__ void k_final(const float* __restrict__ Wsm,
                        const float* __restrict__ scal,
                        float* __restrict__ out) {
    float s = 0.0f;
    for (int i = threadIdx.x; i < (DIM + F_OUT) * HIDE; i += 256) {
        float x = Wsm[i];
        s += x * x;
    }
    s = wave_reduce_sum(s);
    __shared__ float sh[8];
    if ((threadIdx.x & 31) == 0) sh[threadIdx.x >> 5] = s;
    __syncthreads();
    if (threadIdx.x == 0) {
        float l2w = 0.0f;
        for (int i = 0; i < 8; ++i) l2w += sh[i];
        float loss = 0.1f * scal[2] + 0.2f * 0.5f * (scal[0] + scal[1] + l2w);
        out[0] = loss;
    }
}

// ---------------------------------------------------------------------------
extern "C" void kernel_launch(void* const* d_in, const int* in_sizes, int n_in,
                              void* d_out, int out_size, void* d_ws,
                              size_t ws_size, hipStream_t stream) {
    const float* emb   = (const float*)d_in[0];   // embeddings [200000,64]
    const float* ofeat = (const float*)d_in[1];   // other_feat [200000,32]
    const float* f_w   = (const float*)d_in[2];   // [32,64]
    const float* Wsm   = (const float*)d_in[3];   // emb_smaller [128,128]
    const float* bsm   = (const float*)d_in[4];   // [128]
    const float* Wg    = (const float*)d_in[5];   // gru_gate_kernel [256,256]
    const float* bg    = (const float*)d_in[6];   // [256]
    const float* Wc    = (const float*)d_in[7];   // gru_cand_kernel [256,128]
    const float* bc    = (const float*)d_in[8];   // [128]
    const int*   item  = (const int*)d_in[9];     // [256,50]
    const int*   nses  = (const int*)d_in[10];    // [256]
    const int*   tar   = (const int*)d_in[11];    // [256]

    float* out    = (float*)d_out;
    float* logits = out + 1;                      // softmax output [256,200000]

    char* ws = (char*)d_ws;
    unsigned short* inemb_bf = (unsigned short*)ws;                  // 51.2 MB
    unsigned short* uemb_bf  = (unsigned short*)(ws + 51200000);     // 64 KB
    unsigned*       rowmax   = (unsigned*)(ws + 51265536);           // 1 KB
    float*          rowsum   = (float*)(ws + 51266560);              // 1 KB
    float*          scal     = (float*)(ws + 51267584);              // 32 B
    unsigned short* f_wT     = (unsigned short*)(ws + 51267712);     // 4 KB
    unsigned short* WsmT     = (unsigned short*)(ws + 51271808);     // 32 KB
    unsigned short* WgHT     = (unsigned short*)(ws + 51304576);     // 64 KB
    unsigned short* WcHT     = (unsigned short*)(ws + 51370112);     // 32 KB

    k_init<<<1, 256, 0, stream>>>(rowmax, rowsum, scal);
    k_prep<<<128, 256, 0, stream>>>(f_w, Wsm, Wg, Wc, f_wT, WsmT, WgHT, WcHT);
    k_inemb<<<N_NODE / 64, 128, 0, stream>>>(emb, ofeat, f_wT, WsmT, bsm,
                                             inemb_bf, &scal[0]);
    k_gru<<<(BATCH * SEQ) / 64, 128, 0, stream>>>(inemb_bf, item, nses, WgHT,
                                                  bg, WcHT, bc, uemb_bf,
                                                  &scal[1]);
    k_logits<<<dim3(125, 16), 128, 0, stream>>>(uemb_bf, inemb_bf, logits,
                                                rowmax);
    k_expsum<<<dim3(98, BATCH), 256, 0, stream>>>(logits, rowmax, rowsum);
    k_ce<<<1, 256, 0, stream>>>(logits, rowsum, tar, &scal[2]);
    k_norm<<<dim3(98, BATCH), 256, 0, stream>>>(logits, rowsum);
    k_final<<<1, 256, 0, stream>>>(Wsm, scal, out);
}
